// Siren_89326729822766
// MI455X (gfx1250) — matrix operand order repaired and checked by reading, not compiled, not run on hardware
//
#include <hip/hip_runtime.h>

typedef __attribute__((ext_vector_type(16))) _Float16 v16h;
typedef __attribute__((ext_vector_type(2)))  _Float16 h2v;
typedef __attribute__((ext_vector_type(8)))  float    v8f;

#define WAVES          8
#define TILES_PER_WAVE 8
#define HSTRIDE        40   // halfs per H row (32 data + 8 pad) -> 80B stride, 16B aligned

// H tiles are stored with interleaved columns: value for global feature
// n = t*16 + nl (t = N-tile 0/1, nl = lane column) lives at column 2*nl + t.
// So H column k holds feature g(k); the NEXT layer's weight rows are permuted
// by g when the B fragment is built (once per wave) to compensate.
__device__ __forceinline__ int gperm(int k) { return ((k & 1) << 4) | (k >> 1); }

// NeRF frequency encoding element: enc column c (0..47) for point (x0,x1)
//   c in [0,24): dim0 ; c in [24,48): dim1
//   within dim: j<12 -> sin(pi*2^j * x), j>=12 -> cos(pi*2^(j-12) * x)
__device__ __forceinline__ float enc_val(float x0, float x1, int c) {
  const float xv = (c < 24) ? x0 : x1;
  int j = (c < 24) ? c : c - 24;
  const bool is_cos = (j >= 12);
  const int  f = is_cos ? (j - 12) : j;
  const float ph = xv * (3.14159265358979323846f * (float)(1 << f));
  return is_cos ? __cosf(ph) : __sinf(ph);
}

__global__ __launch_bounds__(256)
void siren_ffmlp_kernel(const float* __restrict__ X,
                        const float* __restrict__ W0,
                        const float* __restrict__ W1,
                        const float* __restrict__ W2,
                        float* __restrict__ out, int N)
{
  __shared__ __align__(16) _Float16 Hbuf[WAVES][16 * HSTRIDE];

  const int  lane = threadIdx.x & 31;
  const int  wave = threadIdx.x >> 5;
  const bool lo   = lane < 16;
  const int  ncol = lane & 15;
  const int  kb   = lo ? 0 : 16;  // unused; B fragment is built with ka (same K order as A)
  const int  ka   = lo ? 0 : 8;   // A-fragment K base within low 16 K's
  const int  mC   = lo ? 0 : 8;   // C/D-fragment M base (M=v / M=8+v)

  // ---- Build all B fragments (weights, f32->f16) in registers, once per wave ----
  // B layout (16-bit, 32x16): lane = column N (0..15), halfs 0..7 hold
  // K = ka..ka+7, halfs 8..15 hold K = 16+ka..23+ka (same as A).
  v16h b00, b01, b10, b11, w1a, w1b, b2;
  #pragma unroll
  for (int j = 0; j < 16; ++j) {
    const int k = (j < 8) ? (ka + j) : (16 + ka + (j - 8));
    b00[j] = (_Float16)W0[k * 32 + ncol];               // W0 rows 0..31, cols 0..15
    b01[j] = (_Float16)W0[k * 32 + 16 + ncol];          // W0 rows 0..31, cols 16..31
    const int k1 = 32 + k;                              // W0 rows 32..47, zero-pad to 64
    b10[j] = (_Float16)((k1 < 48) ? W0[k1 * 32 + ncol]      : 0.f);
    b11[j] = (_Float16)((k1 < 48) ? W0[k1 * 32 + 16 + ncol] : 0.f);
    const int gk = gperm(k);                            // undo H column interleave
    w1a[j] = (_Float16)W1[gk * 32 + ncol];
    w1b[j] = (_Float16)W1[gk * 32 + 16 + ncol];
    b2[j]  = (_Float16)((ncol == 0) ? W2[gk] : 0.f);    // 32x16, only col 0 non-zero
  }

  _Float16* Hw = &Hbuf[wave][0];
  const int tile0 = (blockIdx.x * WAVES + wave) * TILES_PER_WAVE;

  for (int t = 0; t < TILES_PER_WAVE; ++t) {
    const int base = (tile0 + t) * 16;
    if (base >= N) break;                         // wave-uniform branch, EXEC stays full

    // ---- frequency-encode straight into A-fragment layout ----
    // A layout (16-bit, 16x32): lane = row M (both halves), halfs 0..7 hold
    // K = ka..ka+7, halfs 8..15 hold K = 16+ka..23+ka.
    const int row = base + ncol;
    float x0 = 0.f, x1 = 0.f;
    if (row < N) {
      const float2 xv = reinterpret_cast<const float2*>(X)[row];
      x0 = xv.x; x1 = xv.y;
    }
    v16h a0, a1;
    #pragma unroll
    for (int i = 0; i < 8; ++i) {
      a0[i]     = (_Float16)enc_val(x0, x1, ka + i);          // enc cols 0..31
      a0[8 + i] = (_Float16)enc_val(x0, x1, 16 + ka + i);
      a1[i]     = (_Float16)enc_val(x0, x1, 32 + ka + i);     // enc cols 32..47
      a1[8 + i] = (_Float16)0.0f;                             // zero-pad K=48..63
    }

    // ---- layer 0: enc[16x48] @ W0[48x32] -> 4 WMMAs ----
    v8f acc0 = {}; v8f acc1 = {};
    acc0 = __builtin_amdgcn_wmma_f32_16x16x32_f16(false, a0, false, b00, (short)0, acc0, false, false);
    acc0 = __builtin_amdgcn_wmma_f32_16x16x32_f16(false, a1, false, b10, (short)0, acc0, false, false);
    acc1 = __builtin_amdgcn_wmma_f32_16x16x32_f16(false, a0, false, b01, (short)0, acc1, false, false);
    acc1 = __builtin_amdgcn_wmma_f32_16x16x32_f16(false, a1, false, b11, (short)0, acc1, false, false);

    // ---- sin activation; packed b32 spill, interleaved columns ----
    #pragma unroll
    for (int v = 0; v < 8; ++v) {
      h2v pk = { (_Float16)__sinf(acc0[v]), (_Float16)__sinf(acc1[v]) };
      *reinterpret_cast<h2v*>(&Hw[(mC + v) * HSTRIDE + 2 * ncol]) = pk;
    }
    asm volatile("s_wait_dscnt 0" ::: "memory");  // wave-local LDS RAW fence

    // ---- reload as A fragment: two b128 loads per lane ----
    union { v16h v; uint4 q[2]; } h1;
    const char* hb = reinterpret_cast<const char*>(Hw)
                     + ncol * (HSTRIDE * 2)        // row m = lane&15, 80B stride
                     + (lo ? 0 : 16);              // K 0..7 vs 8..15
    h1.q[0] = *reinterpret_cast<const uint4*>(hb);        // K 0..7  / 8..15
    h1.q[1] = *reinterpret_cast<const uint4*>(hb + 32);   // K 16..23 / 24..31
    asm volatile("" ::: "memory");

    // ---- layer 1: h[16x32] @ W1'[32x32] (rows g-permuted) -> 2 WMMAs ----
    v8f c0 = {}; v8f c1 = {};
    c0 = __builtin_amdgcn_wmma_f32_16x16x32_f16(false, h1.v, false, w1a, (short)0, c0, false, false);
    c1 = __builtin_amdgcn_wmma_f32_16x16x32_f16(false, h1.v, false, w1b, (short)0, c1, false, false);

    // ---- sin activation; second packed spill (same interleave) ----
    #pragma unroll
    for (int v = 0; v < 8; ++v) {
      h2v pk = { (_Float16)__sinf(c0[v]), (_Float16)__sinf(c1[v]) };
      *reinterpret_cast<h2v*>(&Hw[(mC + v) * HSTRIDE + 2 * ncol]) = pk;
    }
    asm volatile("s_wait_dscnt 0" ::: "memory");

    union { v16h v; uint4 q[2]; } h2;
    h2.q[0] = *reinterpret_cast<const uint4*>(hb);
    h2.q[1] = *reinterpret_cast<const uint4*>(hb + 32);
    asm volatile("" ::: "memory");

    // ---- layer 2: h2[16x32] @ W2'[32x16] (col 0 only) -> 1 WMMA ----
    v8f d = {};
    d = __builtin_amdgcn_wmma_f32_16x16x32_f16(false, h2.v, false, b2, (short)0, d, false, false);

    // D column 0 lives in lanes 0 (M=0..7) and 16 (M=8..15), M contiguous in VGPRs.
    if (ncol == 0) {
      const int r0 = base + mC;
      if (r0 + 7 < N) {
        *reinterpret_cast<float4*>(&out[r0])     = make_float4(d[0], d[1], d[2], d[3]);
        *reinterpret_cast<float4*>(&out[r0 + 4]) = make_float4(d[4], d[5], d[6], d[7]);
      } else {
        #pragma unroll
        for (int v = 0; v < 8; ++v)
          if (r0 + v < N) out[r0 + v] = d[v];
      }
    }
  }
}

extern "C" void kernel_launch(void* const* d_in, const int* in_sizes, int n_in,
                              void* d_out, int out_size, void* d_ws, size_t ws_size,
                              hipStream_t stream) {
  (void)n_in; (void)out_size; (void)d_ws; (void)ws_size;
  const float* X  = (const float*)d_in[0];
  const float* W0 = (const float*)d_in[1];
  const float* W1 = (const float*)d_in[2];
  const float* W2 = (const float*)d_in[3];
  float* out = (float*)d_out;

  const int N = in_sizes[0] / 2;                          // x is [N,2]
  const int pts_per_block = 16 * WAVES * TILES_PER_WAVE;  // 1024
  const int blocks = (N + pts_per_block - 1) / pts_per_block;
  hipLaunchKernelGGL(siren_ffmlp_kernel, dim3(blocks), dim3(256), 0, stream,
                     X, W0, W1, W2, out, N);
}
